// AdaptiveNeighborhoodFusionTransformer_68324339745200
// MI455X (gfx1250) — compile-verified, matrix operand order checked by reading, not run on hardware
//
#include <hip/hip_runtime.h>
#include <hip/hip_bf16.h>

// ---------------------------------------------------------------------------
// MI455X (gfx1250) implementation.
// GEMM-dominated workload (~2.5 TFLOP bf16) on [4096 x 768] token matrices.
// Engine: v_wmma_f32_16x16x32_bf16 tiled GEMM, double-buffered LDS fed by
// GLOBAL_LOAD_ASYNC_TO_LDS_B128 (ASYNCcnt pipeline), branch-free steady state.
// ---------------------------------------------------------------------------

typedef __bf16  bf16x16 __attribute__((ext_vector_type(16)));
typedef float   f32x8   __attribute__((ext_vector_type(8)));
typedef int     i32x4   __attribute__((ext_vector_type(4)));

#define TOK    4096         // B*L  (1024 * 4)
#define DMODEL 768
#define NHEAD  8
#define HDIM   96

#define BM 128
#define BN 128
#define KT 32
#define LDK 40              // padded LDS stride (elems); 80B keeps 16B align + no bank conflicts

enum { ACT_NONE = 0, ACT_GELU = 1, ACT_RELU = 2 };

__device__ __forceinline__ float gelu_f(float x) {
    return 0.5f * x * (1.0f + erff(x * 0.70710678118654752f));
}
__device__ __forceinline__ float sigmoid_f(float x) {
    return 1.0f / (1.0f + expf(-x));
}

// ----------------------- CDNA5 async copy helpers --------------------------
__device__ __forceinline__ void async_b128(void* lds, const void* gaddr) {
#if __has_builtin(__builtin_amdgcn_global_load_async_to_lds_b128)
    __builtin_amdgcn_global_load_async_to_lds_b128(
        (__attribute__((address_space(1))) i32x4*)(void*)gaddr,
        (__attribute__((address_space(3))) i32x4*)lds, 0, 0);
#else
    unsigned loff = (unsigned)(unsigned long long)lds;   // low 32b = LDS offset
    asm volatile("global_load_async_to_lds_b128 %0, %1, off"
                 :: "v"(loff), "v"(gaddr) : "memory");
#endif
}

template <int N>
__device__ __forceinline__ void wait_async() {
#if __has_builtin(__builtin_amdgcn_s_wait_asynccnt)
    __builtin_amdgcn_s_wait_asynccnt((short)N);
#else
    asm volatile("s_wait_asynccnt %0" :: "i"(N));
#endif
}

// --------------------------- f32 -> bf16 cast ------------------------------
__global__ __launch_bounds__(256) void k_cast_bf16(__hip_bfloat16* __restrict__ dst,
                                                   const float* __restrict__ src,
                                                   size_t n) {
    size_t i = (size_t)blockIdx.x * blockDim.x + threadIdx.x;
    if (i < n) dst[i] = __float2bfloat16(src[i]);
}

// --------------------------- WMMA bf16 GEMM --------------------------------
// act(A[M,K] @ W[N,K]^T + bias) -> C (f32, optional) and/or Cbf (bf16, optional).
// M%128==0, N%128==0, K%32==0.
__device__ __forceinline__ bf16x16 load_frag(const __hip_bfloat16* base, int kh) {
    union { bf16x16 v; uint4 u[2]; } f;
    f.u[0] = *(const uint4*)(base + kh);        // K = kh .. kh+7
    f.u[1] = *(const uint4*)(base + kh + 16);   // K = kh+16 .. kh+23
    return f.v;
}

__global__ __launch_bounds__(256)
void k_bf16gemm(const __hip_bfloat16* __restrict__ A,
                const __hip_bfloat16* __restrict__ W,
                const float* __restrict__ bias,
                float* __restrict__ C,
                __hip_bfloat16* __restrict__ Cbf,
                int M, int N, int K, int act) {
    // double-buffered stage: [A 128x32 | B 128x32] bf16, padded stride
    __shared__ __align__(16) __hip_bfloat16 sm[2][(BM + BN) * LDK];

    const int tid  = threadIdx.x;
    const int lane = tid & 31;
    const int wave = tid >> 5;
    const int m0 = blockIdx.x * BM;
    const int n0 = blockIdx.y * BN;
    const int wm = (wave >> 1) * 32;   // 4 waves down M
    const int wn = (wave & 1) * 64;    // 2 waves across N, 64 wide each

    // 4 async b128 chunks per thread per stage (A:2, B:2)
    const int r0 = tid >> 2,          c0 = (tid & 3) * 8;
    const int r1 = (tid + 256) >> 2,  c1 = ((tid + 256) & 3) * 8;

    // running global source pointers (advance by KT per stage)
    const __hip_bfloat16* pa0 = A + (size_t)(m0 + r0) * K + c0;
    const __hip_bfloat16* pa1 = A + (size_t)(m0 + r1) * K + c1;
    const __hip_bfloat16* pw0 = W + (size_t)(n0 + r0) * K + c0;
    const __hip_bfloat16* pw1 = W + (size_t)(n0 + r1) * K + c1;
    // fixed LDS destinations per buffer
    __hip_bfloat16* la0[2]; __hip_bfloat16* la1[2];
    __hip_bfloat16* lb0[2]; __hip_bfloat16* lb1[2];
    for (int b = 0; b < 2; ++b) {
        la0[b] = sm[b] + r0 * LDK + c0;
        la1[b] = sm[b] + r1 * LDK + c1;
        lb0[b] = sm[b] + BM * LDK + r0 * LDK + c0;
        lb1[b] = sm[b] + BM * LDK + r1 * LDK + c1;
    }

    f32x8 acc[2][4] = {};

    auto stage = [&](int buf) {
        async_b128(la0[buf], pa0);  async_b128(la1[buf], pa1);
        async_b128(lb0[buf], pw0);  async_b128(lb1[buf], pw1);
        pa0 += KT; pa1 += KT; pw0 += KT; pw1 += KT;
    };

    const int r  = lane & 15;
    const int kh = (lane >> 4) * 8;          // ISA 16-bit fragment K-half split
    auto compute = [&](int buf) {
        const __hip_bfloat16* As = sm[buf];
        const __hip_bfloat16* Bs = sm[buf] + BM * LDK;
        bf16x16 af[2], bfr[4];
        af[0] = load_frag(As + (wm +  0 + r) * LDK, kh);
        af[1] = load_frag(As + (wm + 16 + r) * LDK, kh);
        for (int j = 0; j < 4; ++j)
            bfr[j] = load_frag(Bs + (wn + j * 16 + r) * LDK, kh);
        for (int i = 0; i < 2; ++i)
            for (int j = 0; j < 4; ++j)
                acc[i][j] = __builtin_amdgcn_wmma_f32_16x16x32_bf16(
                    false, af[i], false, bfr[j], (short)0, acc[i][j], false, false);
    };

    // branch-free steady state: stage(next) -> wait prev stage -> compute
    stage(0);
    int cur = 0;
    for (int k0 = 0; k0 + KT < K; k0 += KT) {
        stage(cur ^ 1);
        wait_async<4>();            // previous stage's 4 loads have landed
        __syncthreads();
        compute(cur);
        __syncthreads();
        cur ^= 1;
    }
    wait_async<0>();                // drain final stage
    __syncthreads();
    compute(cur);

    // C/D layout: VGPR e -> M = e (lanes 0-15) / e+8 (lanes 16-31); N = lane&15
    const int mh = (lane >> 4) * 8;
    for (int i = 0; i < 2; ++i)
        for (int j = 0; j < 4; ++j) {
            const int colg = n0 + wn + j * 16 + r;
            float bv = bias ? bias[colg] : 0.0f;
            for (int e = 0; e < 8; ++e) {
                int rowg = m0 + wm + i * 16 + mh + e;
                float v = acc[i][j][e] + bv;
                if (act == ACT_GELU)      v = gelu_f(v);
                else if (act == ACT_RELU) v = fmaxf(v, 0.0f);
                if (C)   C  [(size_t)rowg * N + colg] = v;
                if (Cbf) Cbf[(size_t)rowg * N + colg] = __float2bfloat16(v);
            }
        }
}

// ------------------------- layout / elementwise ----------------------------
__global__ __launch_bounds__(256) void k_nchw_tok(float* __restrict__ dst,
                                                  const float* __restrict__ src, size_t n) {
    size_t i = (size_t)blockIdx.x * blockDim.x + threadIdx.x;
    if (i >= n) return;
    int d = (int)(i % DMODEL);
    size_t row = i / DMODEL;
    size_t b = row >> 2; int l = (int)(row & 3);
    dst[i] = src[(b * DMODEL + d) * 4 + l];
}

__global__ __launch_bounds__(256) void k_spatial_update(float* __restrict__ f,
                                                        const float* __restrict__ gp,
                                                        const float* __restrict__ pos, size_t n) {
    size_t i = (size_t)blockIdx.x * blockDim.x + threadIdx.x;
    if (i >= n) return;
    int d = (int)(i % DMODEL);
    int l = (int)((i / DMODEL) & 3);
    f[i] += pos[l * DMODEL + d] * sigmoid_f(gp[i]);
}

__global__ __launch_bounds__(256) void k_add(float* __restrict__ dst, const float* __restrict__ a,
                                             const float* __restrict__ b, size_t n) {
    size_t i = (size_t)blockIdx.x * blockDim.x + threadIdx.x;
    if (i < n) dst[i] = a[i] + b[i];
}

// dst = base + tanh(s)*delta ; optional bf16 mirror for the next GEMM input
__global__ __launch_bounds__(256) void k_axpy_tanh(float* __restrict__ dst,
                                                   __hip_bfloat16* __restrict__ dbf,
                                                   const float* __restrict__ base,
                                                   const float* __restrict__ delta,
                                                   const float* __restrict__ sptr, size_t n) {
    size_t i = (size_t)blockIdx.x * blockDim.x + threadIdx.x;
    if (i >= n) return;
    float v = base[i] + tanhf(sptr[0]) * delta[i];
    dst[i] = v;
    if (dbf) dbf[i] = __float2bfloat16(v);
}

__global__ __launch_bounds__(256) void k_scale_acc(float* __restrict__ dst, const float* __restrict__ src,
                                                   size_t n, int first) {
    size_t i = (size_t)blockIdx.x * blockDim.x + threadIdx.x;
    if (i >= n) return;
    float v = 0.5f * src[i];
    dst[i] = first ? v : dst[i] + v;
}

// softmax(pool_w)-weighted concat into [TOK, 4D]
__global__ __launch_bounds__(256) void k_pool_concat(float* __restrict__ dst,
                                                     const float* __restrict__ f0, const float* __restrict__ f1,
                                                     const float* __restrict__ f2, const float* __restrict__ f3,
                                                     const float* __restrict__ pw, size_t n) {
    size_t i = (size_t)blockIdx.x * blockDim.x + threadIdx.x;
    if (i >= n) return;
    int d = (int)(i % DMODEL);
    size_t row = i / DMODEL;
    float w0 = pw[0], w1 = pw[1], w2 = pw[2], w3 = pw[3];
    float m = fmaxf(fmaxf(w0, w1), fmaxf(w2, w3));
    float e0 = expf(w0 - m), e1 = expf(w1 - m), e2 = expf(w2 - m), e3 = expf(w3 - m);
    float inv = 1.0f / (e0 + e1 + e2 + e3);
    size_t o = row * (4 * DMODEL) + d;
    dst[o + 0 * DMODEL] = f0[i] * e0 * inv;
    dst[o + 1 * DMODEL] = f1[i] * e1 * inv;
    dst[o + 2 * DMODEL] = f2[i] * e2 * inv;
    dst[o + 3 * DMODEL] = f3[i] * e3 * inv;
}

// Multiscale grouped Conv1d (groups=96, io=8, k in {1,2,4}) + BN(eval) + GELU,
// concatenated to [TOK, 3D].
__global__ __launch_bounds__(256)
void k_msconv(float* __restrict__ out, const float* __restrict__ x,
              const float* __restrict__ w1, const float* __restrict__ b1,
              const float* __restrict__ g1, const float* __restrict__ bb1,
              const float* __restrict__ w2, const float* __restrict__ b2,
              const float* __restrict__ g2, const float* __restrict__ bb2,
              const float* __restrict__ w4, const float* __restrict__ b4,
              const float* __restrict__ g4, const float* __restrict__ bb4, size_t n) {
    size_t i = (size_t)blockIdx.x * blockDim.x + threadIdx.x;
    if (i >= n) return;
    int d = (int)(i % DMODEL);
    size_t row = i / DMODEL;
    size_t b = row >> 2; int l = (int)(row & 3);
    int grp = d >> 3;
    const float bnscale = rsqrtf(1.0f + 1e-5f);
    size_t xb = (b * 4) * DMODEL + grp * 8;
    size_t ob = row * (3 * DMODEL) + d;

    { float a = 0.f;                                   // k=1, pad 0
      for (int j = 0; j < 8; ++j) a += w1[d * 8 + j] * x[xb + (size_t)l * DMODEL + j];
      a = (a + b1[d]) * bnscale * g1[d] + bb1[d];
      out[ob] = gelu_f(a); }
    { float a = 0.f;                                   // k=2, pad 1, crop
      for (int t = 0; t < 2; ++t) {
          int li = l - 1 + t;
          if (li < 0 || li > 3) continue;
          for (int j = 0; j < 8; ++j) a += w2[(d * 8 + j) * 2 + t] * x[xb + (size_t)li * DMODEL + j];
      }
      a = (a + b2[d]) * bnscale * g2[d] + bb2[d];
      out[ob + DMODEL] = gelu_f(a); }
    { float a = 0.f;                                   // k=4, pad 2, crop
      for (int t = 0; t < 4; ++t) {
          int li = l - 2 + t;
          if (li < 0 || li > 3) continue;
          for (int j = 0; j < 8; ++j) a += w4[(d * 8 + j) * 4 + t] * x[xb + (size_t)li * DMODEL + j];
      }
      a = (a + b4[d]) * bnscale * g4[d] + bb4[d];
      out[ob + 2 * DMODEL] = gelu_f(a); }
}

// -------------------------- LayerNorm family -------------------------------
__device__ __forceinline__ void wave_red2(float& s, float& ss) {
    for (int o = 16; o; o >>= 1) { s += __shfl_xor(s, o, 32); ss += __shfl_xor(ss, o, 32); }
}

__global__ __launch_bounds__(256) void k_ln(float* __restrict__ out, const float* __restrict__ in,
                                            const float* __restrict__ g, const float* __restrict__ b,
                                            int rows, int D) {
    int gw = (int)((blockIdx.x * (size_t)blockDim.x + threadIdx.x) >> 5);
    int lane = threadIdx.x & 31;
    if (gw >= rows) return;
    const float* x = in + (size_t)gw * D;
    float s = 0.f, ss = 0.f;
    for (int d = lane; d < D; d += 32) { float v = x[d]; s += v; ss += v * v; }
    wave_red2(s, ss);
    float m = s / D, inv = rsqrtf(ss / D - m * m + 1e-5f);
    float* y = out + (size_t)gw * D;
    for (int d = lane; d < D; d += 32) y[d] = (x[d] - m) * inv * g[d] + b[d];
}

// LN with bf16 output (feeds a GEMM directly)
__global__ __launch_bounds__(256) void k_ln_bf(__hip_bfloat16* __restrict__ out,
                                               const float* __restrict__ in,
                                               const float* __restrict__ g, const float* __restrict__ b,
                                               int rows, int D) {
    int gw = (int)((blockIdx.x * (size_t)blockDim.x + threadIdx.x) >> 5);
    int lane = threadIdx.x & 31;
    if (gw >= rows) return;
    const float* x = in + (size_t)gw * D;
    float s = 0.f, ss = 0.f;
    for (int d = lane; d < D; d += 32) { float v = x[d]; s += v; ss += v * v; }
    wave_red2(s, ss);
    float m = s / D, inv = rsqrtf(ss / D - m * m + 1e-5f);
    __hip_bfloat16* y = out + (size_t)gw * D;
    for (int d = lane; d < D; d += 32) y[d] = __float2bfloat16((x[d] - m) * inv * g[d] + b[d]);
}

__global__ __launch_bounds__(256) void k_gate_ln(float* __restrict__ out, const float* __restrict__ o,
                                                 const float* __restrict__ enh, const float* __restrict__ gp,
                                                 const float* __restrict__ g, const float* __restrict__ b,
                                                 int rows, int D) {
    int gw = (int)((blockIdx.x * (size_t)blockDim.x + threadIdx.x) >> 5);
    int lane = threadIdx.x & 31;
    if (gw >= rows) return;
    size_t base = (size_t)gw * D;
    float s = 0.f, ss = 0.f;
    for (int d = lane; d < D; d += 32) {
        float t = o[base + d] + enh[base + d] * sigmoid_f(gp[base + d]);
        s += t; ss += t * t;
    }
    wave_red2(s, ss);
    float m = s / D, inv = rsqrtf(ss / D - m * m + 1e-5f);
    for (int d = lane; d < D; d += 32) {
        float t = o[base + d] + enh[base + d] * sigmoid_f(gp[base + d]);
        out[base + d] = (t - m) * inv * g[d] + b[d];   // in-place safe: lane-owned elems
    }
}

// final = LN(g + attn), written back to NCHW (B, D, 2, 2)
__global__ __launch_bounds__(256) void k_final_ln_tr(float* __restrict__ out, const float* __restrict__ gin,
                                                     const float* __restrict__ attn,
                                                     const float* __restrict__ g, const float* __restrict__ b,
                                                     int rows) {
    int gw = (int)((blockIdx.x * (size_t)blockDim.x + threadIdx.x) >> 5);
    int lane = threadIdx.x & 31;
    if (gw >= rows) return;
    size_t base = (size_t)gw * DMODEL;
    size_t bb = (size_t)(gw >> 2); int l = gw & 3;
    float s = 0.f, ss = 0.f;
    for (int d = lane; d < DMODEL; d += 32) {
        float t = gin[base + d] + attn[base + d];
        s += t; ss += t * t;
    }
    wave_red2(s, ss);
    float m = s / DMODEL, inv = rsqrtf(ss / DMODEL - m * m + 1e-5f);
    for (int d = lane; d < DMODEL; d += 32) {
        float t = gin[base + d] + attn[base + d];
        out[(bb * DMODEL + d) * 4 + l] = (t - m) * inv * g[d] + b[d];
    }
}

// ----------------------- L=4 attention (wave per b,h) ----------------------
__global__ __launch_bounds__(256)
void k_attn4(const float* __restrict__ Q, const float* __restrict__ KV,
             float* __restrict__ O, int ld, int out_ld, int out_off,
             const float* __restrict__ tptr, int npairs) {
    int gw = (int)((blockIdx.x * (size_t)blockDim.x + threadIdx.x) >> 5);
    int lane = threadIdx.x & 31;
    if (gw >= npairs) return;
    int b = gw >> 3, h = gw & 7;
    int qb = h * HDIM, kb = DMODEL + h * HDIM, vb = 2 * DMODEL + h * HDIM;
    float q[4][3], k[4][3], v[4][3];
    for (int l = 0; l < 4; ++l) {
        size_t r = (size_t)(b * 4 + l) * ld;
        for (int j = 0; j < 3; ++j) {
            int d = lane * 3 + j;
            q[l][j] = Q[r + qb + d];
            k[l][j] = KV[r + kb + d];
            v[l][j] = KV[r + vb + d];
        }
    }
    float t = 1.0f;
    if (tptr) { t = tptr[0]; t = t < 0.01f ? 0.01f : t; }
    float sc = rsqrtf((float)HDIM) / t;
    float p[4][4];
    for (int i = 0; i < 4; ++i)
        for (int j = 0; j < 4; ++j) {
            float s = q[i][0] * k[j][0] + q[i][1] * k[j][1] + q[i][2] * k[j][2];
            for (int o = 16; o; o >>= 1) s += __shfl_xor(s, o, 32);
            p[i][j] = s * sc;
        }
    for (int i = 0; i < 4; ++i) {
        float m = fmaxf(fmaxf(p[i][0], p[i][1]), fmaxf(p[i][2], p[i][3]));
        float e0 = expf(p[i][0] - m), e1 = expf(p[i][1] - m), e2 = expf(p[i][2] - m), e3 = expf(p[i][3] - m);
        float inv = 1.0f / (e0 + e1 + e2 + e3);
        p[i][0] = e0 * inv; p[i][1] = e1 * inv; p[i][2] = e2 * inv; p[i][3] = e3 * inv;
    }
    for (int i = 0; i < 4; ++i) {
        size_t r = (size_t)(b * 4 + i) * out_ld + out_off + h * HDIM;
        for (int j = 0; j < 3; ++j) {
            int d = lane * 3 + j;
            O[r + d] = p[i][0] * v[0][j] + p[i][1] * v[1][j] + p[i][2] * v[2][j] + p[i][3] * v[3][j];
        }
    }
}

// ============================ host orchestration ===========================
struct Lin { const float* w; const float* b; };
struct CAp {
    const float *na_g, *na_b, *nb_g, *nb_b;
    const float *wq_a, *wk_a, *wv_a, *wq_b, *wk_b, *wv_b;
    const float *temp, *alpha, *beta;
    Lin cf1, cf2, ffn1, ffn2, outp;
};
struct MSc { const float *w, *b, *g, *bb; };
struct Lyr { CAp ca12, ca14, ca23, ca34; const float *ln_g, *ln_b; Lin gate; };

struct Ctx {
    hipStream_t s;
    __hip_bfloat16 *abf, *bbf, *wbf;
};

static inline int cdiv_i(size_t a, int b) { return (int)((a + b - 1) / b); }

static void castb(Ctx& c, __hip_bfloat16* dst, const float* src, size_t n) {
    k_cast_bf16<<<cdiv_i(n, 256), 256, 0, c.s>>>(dst, src, n);
}
static void gemm_core(Ctx& c, const __hip_bfloat16* A, const __hip_bfloat16* W,
                      const float* bias, float* C, __hip_bfloat16* Cbf,
                      int M, int N, int K, int act) {
    dim3 g(M / BM, N / BN);
    k_bf16gemm<<<g, 256, 0, c.s>>>(A, W, bias, C, Cbf, M, N, K, act);
}
// generic: f32 A, f32 W -> outputs
static void gemm(Ctx& c, const float* A, int M, int K, Lin lin, int N,
                 float* C, __hip_bfloat16* Cbf, int act) {
    castb(c, c.abf, A, (size_t)M * K);
    castb(c, c.wbf, lin.w, (size_t)N * K);
    gemm_core(c, c.abf, c.wbf, lin.b, C, Cbf, M, N, K, act);
}
static void ln(Ctx& c, float* out, const float* in, const float* g, const float* b, int rows, int D) {
    k_ln<<<cdiv_i((size_t)rows * 32, 256), 256, 0, c.s>>>(out, in, g, b, rows, D);
}

static void cross_attn(Ctx& c, const CAp& p, const float* xa, const float* xb,
                       float* tA, float* tB, float* tC, float* tD, float* tE, float* tF) {
    const int M = TOK, D = DMODEL;
    const size_t S1 = (size_t)M * D;
    const size_t DD = (size_t)D * D;
    const int lnb = cdiv_i((size_t)M * 32, 256);
    const int eb  = cdiv_i(S1, 256);
    // an = LN(xa) -> bf16 ; qkv_a = an @ [Wq;Wk;Wv]^T (one N=2304 GEMM)
    k_ln_bf<<<lnb, 256, 0, c.s>>>(c.abf, xa, p.na_g, p.na_b, M, D);
    castb(c, c.wbf + 0 * DD, p.wq_a, DD);
    castb(c, c.wbf + 1 * DD, p.wk_a, DD);
    castb(c, c.wbf + 2 * DD, p.wv_a, DD);
    gemm_core(c, c.abf, c.wbf, nullptr, tA, nullptr, M, 3 * D, D, ACT_NONE);
    k_ln_bf<<<lnb, 256, 0, c.s>>>(c.abf, xb, p.nb_g, p.nb_b, M, D);
    castb(c, c.wbf + 0 * DD, p.wq_b, DD);
    castb(c, c.wbf + 1 * DD, p.wk_b, DD);
    castb(c, c.wbf + 2 * DD, p.wv_b, DD);
    gemm_core(c, c.abf, c.wbf, nullptr, tB, nullptr, M, 3 * D, D, ACT_NONE);
    // oa (qa x kb,vb) -> tC[:, :768] ; ob (qb x ka,va) -> tC[:, 768:]
    const int npairs = (TOK / 4) * NHEAD;
    const int ab = cdiv_i((size_t)npairs * 32, 256);
    k_attn4<<<ab, 256, 0, c.s>>>(tA, tB, tC, 3 * D, 2 * D, 0, p.temp, npairs);
    k_attn4<<<ab, 256, 0, c.s>>>(tB, tA, tC, 3 * D, 2 * D, D, p.temp, npairs);
    // cross-fusion MLP: hidden stays bf16 (cf1 -> bbf -> cf2)
    castb(c, c.abf, tC, (size_t)M * 2 * D);
    castb(c, c.wbf, p.cf1.w, (size_t)4 * D * 2 * D);
    gemm_core(c, c.abf, c.wbf, p.cf1.b, nullptr, c.bbf, M, 4 * D, 2 * D, ACT_GELU);
    castb(c, c.wbf, p.cf2.w, (size_t)D * 4 * D);
    gemm_core(c, c.bbf, c.wbf, p.cf2.b, tD, nullptr, M, D, 4 * D, ACT_NONE);
    k_axpy_tanh<<<eb, 256, 0, c.s>>>(tE, c.abf, xa, tD, p.alpha, S1);          // xe (f32 + bf16)
    // FFN: ffn1 -> bbf -> ffn2
    castb(c, c.wbf, p.ffn1.w, (size_t)4 * D * D);
    gemm_core(c, c.abf, c.wbf, p.ffn1.b, nullptr, c.bbf, M, 4 * D, D, ACT_GELU);
    castb(c, c.wbf, p.ffn2.w, (size_t)D * 4 * D);
    gemm_core(c, c.bbf, c.wbf, p.ffn2.b, tD, nullptr, M, D, 4 * D, ACT_NONE);
    k_axpy_tanh<<<eb, 256, 0, c.s>>>(tE, c.abf, tE, tD, p.beta, S1);
    castb(c, c.wbf, p.outp.w, DD);
    gemm_core(c, c.abf, c.wbf, p.outp.b, tF, nullptr, M, D, D, ACT_NONE);      // result -> tF
}

extern "C" void kernel_launch(void* const* d_in, const int* in_sizes, int n_in,
                              void* d_out, int out_size, void* d_ws, size_t ws_size,
                              hipStream_t stream) {
    (void)in_sizes; (void)n_in; (void)out_size; (void)ws_size;
    // ---- parse flattened inputs (setup_inputs dict order, depth-first) ----
    int pi = 0;
    auto F = [&]() { return (const float*)d_in[pi++]; };
    const float* fin[4]; for (int i = 0; i < 4; ++i) fin[i] = F();
    const float* pos_embed = F();
    Lin sp1{F(), F()}, sp2{F(), F()};
    MSc msc[3]; for (int i = 0; i < 3; ++i) { msc[i].w = F(); msc[i].b = F(); msc[i].g = F(); msc[i].bb = F(); }
    Lin ms_f1{F(), F()}, ms_f2{F(), F()};
    Lyr lyr[2];
    auto rd_ca = [&](CAp& a) {
        a.na_g = F(); a.na_b = F(); a.nb_g = F(); a.nb_b = F();
        a.wq_a = F(); a.wk_a = F(); a.wv_a = F();
        a.wq_b = F(); a.wk_b = F(); a.wv_b = F();
        a.temp = F(); a.alpha = F(); a.beta = F();
        a.cf1 = {F(), F()}; a.cf2 = {F(), F()};
        a.ffn1 = {F(), F()}; a.ffn2 = {F(), F()}; a.outp = {F(), F()};
    };
    for (int l = 0; l < 2; ++l) {
        rd_ca(lyr[l].ca12); rd_ca(lyr[l].ca14); rd_ca(lyr[l].ca23); rd_ca(lyr[l].ca34);
        lyr[l].ln_g = F(); lyr[l].ln_b = F(); lyr[l].gate = {F(), F()};
    }
    const float* gf_ln1_g = F(); const float* gf_ln1_b = F();
    Lin gf1{F(), F()}, gf2{F(), F()};
    const float* gf_ln2_g = F(); const float* gf_ln2_b = F();
    Lin attn_in{F(), F()}, attn_out{F(), F()};
    const float* sa_g = F(); const float* sa_b = F();
    const float* pool_w = F();

    // ---- workspace arena ----
    const size_t S1 = (size_t)TOK * DMODEL;      // 3.14M elems
    const size_t S4 = (size_t)TOK * 4 * DMODEL;  // 12.58M elems
    char* wp = (char*)d_ws;
    auto arena = [&](size_t bytes) { void* p = wp; wp += (bytes + 255) & ~(size_t)255; return p; };
    Ctx c; c.s = stream;
    c.abf = (__hip_bfloat16*)arena(S4 * sizeof(__hip_bfloat16));   // GEMM input staging
    c.bbf = (__hip_bfloat16*)arena(S4 * sizeof(__hip_bfloat16));   // bf16 hidden chain
    c.wbf = (__hip_bfloat16*)arena((size_t)4 * DMODEL * 2 * DMODEL * sizeof(__hip_bfloat16)); // max W 3072x1536
    float* tA = (float*)arena(S4 * sizeof(float));
    float* tB = (float*)arena(S4 * sizeof(float));
    float* tC = (float*)arena((size_t)TOK * 2 * DMODEL * sizeof(float));
    float* tD = (float*)arena(S1 * sizeof(float));
    float* tE = (float*)arena(S1 * sizeof(float));
    float* tF = (float*)arena(S1 * sizeof(float));
    float* cur[4]; float* nxt[4];
    for (int i = 0; i < 4; ++i) cur[i] = (float*)arena(S1 * sizeof(float));
    for (int i = 0; i < 4; ++i) nxt[i] = (float*)arena(S1 * sizeof(float));

    const int M = TOK, D = DMODEL;
    const int eb  = cdiv_i(S1, 256);
    const int lnb = cdiv_i((size_t)M * 32, 256);

    // ---- 0) NCHW -> tokens ----
    for (int i = 0; i < 4; ++i)
        k_nchw_tok<<<eb, 256, 0, stream>>>(cur[i], fin[i], S1);

    // ---- 1) spatial relation: f += pos[i] * sigmoid(sp2(relu(sp1(f)))) ----
    for (int i = 0; i < 4; ++i) {
        castb(c, c.abf, cur[i], S1);
        castb(c, c.wbf, sp1.w, (size_t)(D / 2) * D);
        gemm_core(c, c.abf, c.wbf, sp1.b, nullptr, c.bbf, M, D / 2, D, ACT_RELU);
        castb(c, c.wbf, sp2.w, (size_t)D * (D / 2));
        gemm_core(c, c.bbf, c.wbf, sp2.b, tD, nullptr, M, D, D / 2, ACT_NONE);
        k_spatial_update<<<eb, 256, 0, stream>>>(cur[i], tD, pos_embed + (size_t)i * 4 * D, S1);
    }

    // ---- 2) multiscale fusion ----
    for (int i = 0; i < 4; ++i) {
        k_msconv<<<eb, 256, 0, stream>>>(tA, cur[i],
                                         msc[0].w, msc[0].b, msc[0].g, msc[0].bb,
                                         msc[1].w, msc[1].b, msc[1].g, msc[1].bb,
                                         msc[2].w, msc[2].b, msc[2].g, msc[2].bb, S1);
        castb(c, c.abf, tA, (size_t)M * 3 * D);
        castb(c, c.wbf, ms_f1.w, (size_t)2 * D * 3 * D);
        gemm_core(c, c.abf, c.wbf, ms_f1.b, nullptr, c.bbf, M, 2 * D, 3 * D, ACT_GELU);
        castb(c, c.wbf, ms_f2.w, (size_t)D * 2 * D);
        gemm_core(c, c.bbf, c.wbf, ms_f2.b, tD, nullptr, M, D, 2 * D, ACT_NONE);
        k_add<<<eb, 256, 0, stream>>>(cur[i], tD, cur[i], S1);
    }

    // ---- 3) bidirectional cross-attention layers ----
    for (int li = 0; li < 2; ++li) {
        const Lyr& L = lyr[li];
        struct Call { const CAp* p; int a, b, dst; };
        const Call calls[8] = {
            {&L.ca12, 0, 1, 0}, {&L.ca14, 0, 3, 0},
            {&L.ca12, 1, 0, 1}, {&L.ca23, 1, 2, 1},
            {&L.ca23, 2, 1, 2}, {&L.ca34, 2, 3, 2},
            {&L.ca14, 3, 0, 3}, {&L.ca34, 3, 2, 3},
        };
        int first[4] = {1, 1, 1, 1};
        for (int k = 0; k < 8; ++k) {
            cross_attn(c, *calls[k].p, cur[calls[k].a], cur[calls[k].b], tA, tB, tC, tD, tE, tF);
            k_scale_acc<<<eb, 256, 0, stream>>>(nxt[calls[k].dst], tF, S1, first[calls[k].dst]);
            first[calls[k].dst] = 0;
        }
        for (int i = 0; i < 4; ++i) {
            castb(c, c.abf, nxt[i], S1);
            castb(c, c.wbf, L.gate.w, (size_t)D * D);
            gemm_core(c, c.abf, c.wbf, L.gate.b, tD, nullptr, M, D, D, ACT_NONE);
            k_gate_ln<<<lnb, 256, 0, stream>>>(cur[i], cur[i], nxt[i], tD, L.ln_g, L.ln_b, M, D);
        }
    }

    // ---- 4) global fusion + self-attention head ----
    k_pool_concat<<<eb, 256, 0, stream>>>(tA, cur[0], cur[1], cur[2], cur[3], pool_w, S1);
    k_ln_bf<<<lnb, 256, 0, stream>>>(c.abf, tA, gf_ln1_g, gf_ln1_b, M, 4 * D);
    castb(c, c.wbf, gf1.w, (size_t)2 * D * 4 * D);
    gemm_core(c, c.abf, c.wbf, gf1.b, nullptr, c.bbf, M, 2 * D, 4 * D, ACT_GELU);
    castb(c, c.wbf, gf2.w, (size_t)D * 2 * D);
    gemm_core(c, c.bbf, c.wbf, gf2.b, tD, nullptr, M, D, 2 * D, ACT_NONE);
    ln(c, tE, tD, gf_ln2_g, gf_ln2_b, M, D);                        // g (f32)
    castb(c, c.abf, tE, S1);
    castb(c, c.wbf, attn_in.w, (size_t)3 * D * D);
    gemm_core(c, c.abf, c.wbf, attn_in.b, tA, nullptr, M, 3 * D, D, ACT_NONE); // qkv
    const int npairs = (TOK / 4) * NHEAD;
    k_attn4<<<cdiv_i((size_t)npairs * 32, 256), 256, 0, stream>>>(tA, tA, tF, 3 * D, D, 0, nullptr, npairs);
    castb(c, c.abf, tF, S1);
    castb(c, c.wbf, attn_out.w, (size_t)D * D);
    gemm_core(c, c.abf, c.wbf, attn_out.b, tD, nullptr, M, D, D, ACT_NONE);    // attn
    k_final_ln_tr<<<lnb, 256, 0, stream>>>((float*)d_out, tE, tD, sa_g, sa_b, M);
}
